// Lstm_10557029614123
// MI455X (gfx1250) — compile-verified, matrix operand order checked by reading, not run on hardware
//
#include <hip/hip_runtime.h>
#include <hip/hip_bf16.h>
#include <math.h>

// ---------------- types ----------------
typedef __bf16 bf16_t;
typedef __attribute__((ext_vector_type(4)))  bf16_t v4bf;
typedef __attribute__((ext_vector_type(8)))  bf16_t v8bf;
typedef __attribute__((ext_vector_type(16))) bf16_t v16bf;
typedef __attribute__((ext_vector_type(8)))  float  v8f;

__device__ __forceinline__ v16bf mk16(v8bf lo, v8bf hi) {
  v16bf r;
#pragma unroll
  for (int i = 0; i < 8; ++i) { r[i] = lo[i]; r[i + 8] = hi[i]; }
  return r;
}
__device__ __forceinline__ float sigmoidf_(float x) {
  return 1.0f / (1.0f + __expf(-x));
}

// ---------------- CDNA5 async copy global -> LDS (ASYNCcnt path) ----------------
// LDS byte offset = low 32 bits of the generic pointer (flat->LDS aperture
// truncation, ISA 10.2).
__device__ __forceinline__ void async_load_b128(const void* gsrc, void* lds_dst) {
  unsigned l = (unsigned)(uintptr_t)lds_dst;
  unsigned long long g = (unsigned long long)(uintptr_t)gsrc;
  asm volatile("global_load_async_to_lds_b128 %0, %1, off"
               :: "v"(l), "v"(g) : "memory");
}
__device__ __forceinline__ void wait_async0() {
#if __has_builtin(__builtin_amdgcn_s_wait_asynccnt)
  __builtin_amdgcn_s_wait_asynccnt(0);
#else
  asm volatile("s_wait_asynccnt 0x0" ::: "memory");
#endif
}

// scheduling-group masks: 0x008=MFMA/WMMA, 0x020=VMEM read, 0x100=DS read
#if __has_builtin(__builtin_amdgcn_sched_group_barrier)
#define SCHED_GROUP(mask, n) __builtin_amdgcn_sched_group_barrier((mask), (n), 0)
#else
#define SCHED_GROUP(mask, n)
#endif

// ---------------- f32 -> bf16 flat convert ----------------
__global__ __launch_bounds__(256) void k_cvt(const float* __restrict__ src,
                                             bf16_t* __restrict__ dst,
                                             long long n) {
  long long i = (long long)blockIdx.x * blockDim.x + threadIdx.x;
  long long stride = (long long)gridDim.x * blockDim.x;
  for (long long j = i * 4; j < n; j += stride * 4) {
    float4 v = *(const float4*)(src + j);
    v4bf o;
    o[0] = (bf16_t)v.x; o[1] = (bf16_t)v.y; o[2] = (bf16_t)v.z; o[3] = (bf16_t)v.w;
    *(v4bf*)(dst + j) = o;
  }
}

// ---------------- f32 -> bf16 transpose: src (R,C) -> dst (C,R) ----------------
__global__ __launch_bounds__(256) void k_transpose_cvt(const float* __restrict__ src,
                                                       bf16_t* __restrict__ dst,
                                                       int R, int C) {
  __shared__ float tile[32][33];
  int c0 = blockIdx.x * 32, r0 = blockIdx.y * 32;
#pragma unroll
  for (int dy = 0; dy < 32; dy += 8) {
    int r = r0 + threadIdx.y + dy, c = c0 + threadIdx.x;
    if (r < R && c < C) tile[threadIdx.y + dy][threadIdx.x] = src[(size_t)r * C + c];
  }
  __syncthreads();
#pragma unroll
  for (int dy = 0; dy < 32; dy += 8) {
    int c = c0 + threadIdx.y + dy, r = r0 + threadIdx.x;
    if (r < R && c < C) dst[(size_t)c * R + r] = (bf16_t)tile[threadIdx.x][threadIdx.y + dy];
  }
}

// ---------------- generic bf16 WMMA GEMM, async double-buffered ----------------
// C(M,N) = A(M,K) * Bt(N,K)^T ; tiles 128x128x32 ; 256 thr = 8 waves (wave32)
// wave w: M-tiles [(w&3)*2,+2), N-tiles [(w>>2)*4,+4)
template <bool OUT_BF16, bool HAS_BIAS>
__global__ __launch_bounds__(256) void k_gemm(const bf16_t* __restrict__ A,
                                              const bf16_t* __restrict__ Bt,
                                              void* __restrict__ Cout,
                                              const float* __restrict__ bias,
                                              int M, int N, int K) {
  const int rowBase = blockIdx.y * 128;
  const int colBase = blockIdx.x * 128;
  const int tid = threadIdx.x, lane = tid & 31, wave = tid >> 5;
  const int mt0 = (wave & 3) * 2, nt0 = (wave >> 2) * 4;
  const int lr = lane & 15;
  const int kb = (lane < 16) ? 0 : 8;   // per-lane K base (ISA A/B bf16 layout)
  const int lrow = tid >> 1;
  const int lk = (tid & 1) * 16;

  __shared__ __align__(16) bf16_t As[2][128 * 32];
  __shared__ __align__(16) bf16_t Bs[2][128 * 32];

  const bf16_t* gA = A + (size_t)(rowBase + lrow) * K + lk;
  const bf16_t* gB = Bt + (size_t)(colBase + lrow) * K + lk;

  v8f zero = {0.f, 0.f, 0.f, 0.f, 0.f, 0.f, 0.f, 0.f};
  v8f acc[2][4];
#pragma unroll
  for (int mi = 0; mi < 2; ++mi)
#pragma unroll
    for (int ni = 0; ni < 4; ++ni) acc[mi][ni] = zero;

  // prologue: async-fetch tile 0
  async_load_b128(gA, &As[0][lrow * 32 + lk]);
  async_load_b128(gA + 8, &As[0][lrow * 32 + lk + 8]);
  async_load_b128(gB, &Bs[0][lrow * 32 + lk]);
  async_load_b128(gB + 8, &Bs[0][lrow * 32 + lk + 8]);

  const int nk = K >> 5;
  for (int ki = 0; ki < nk; ++ki) {
    const int cur = ki & 1;
    wait_async0();        // our async writes to LDS landed
    __syncthreads();      // everyone's landed; prev-iter readers are done
    if (ki + 1 < nk) {    // overlap fetch of tile ki+1 with compute of tile ki
      int k0 = (ki + 1) << 5;
      async_load_b128(gA + k0, &As[cur ^ 1][lrow * 32 + lk]);
      async_load_b128(gA + k0 + 8, &As[cur ^ 1][lrow * 32 + lk + 8]);
      async_load_b128(gB + k0, &Bs[cur ^ 1][lrow * 32 + lk]);
      async_load_b128(gB + k0 + 8, &Bs[cur ^ 1][lrow * 32 + lk + 8]);
    }

    v16bf afr[2], bfr[4];
#pragma unroll
    for (int mi = 0; mi < 2; ++mi) {
      const bf16_t* p = &As[cur][((mt0 + mi) * 16 + lr) * 32 + kb];
      afr[mi] = mk16(*(const v8bf*)p, *(const v8bf*)(p + 16));
    }
#pragma unroll
    for (int ni = 0; ni < 4; ++ni) {
      const bf16_t* p = &Bs[cur][((nt0 + ni) * 16 + lr) * 32 + kb];
      bfr[ni] = mk16(*(const v8bf*)p, *(const v8bf*)(p + 16));
    }
#pragma unroll
    for (int mi = 0; mi < 2; ++mi)
#pragma unroll
      for (int ni = 0; ni < 4; ++ni)
        acc[mi][ni] = __builtin_amdgcn_wmma_f32_16x16x32_bf16(
            false, afr[mi], false, bfr[ni], (short)0, acc[mi][ni], false, false);

    // pin schedule: all 12 LDS fragment reads first, then 8 WMMAs back-to-back
    SCHED_GROUP(0x100, 12);
    SCHED_GROUP(0x008, 8);
  }

  const int rlo = (lane < 16) ? 0 : 8;
#pragma unroll
  for (int mi = 0; mi < 2; ++mi) {
#pragma unroll
    for (int ni = 0; ni < 4; ++ni) {
      int col = colBase + (nt0 + ni) * 16 + lr;
      float bv = HAS_BIAS ? bias[col] : 0.0f;
#pragma unroll
      for (int v = 0; v < 8; ++v) {
        int row = rowBase + (mt0 + mi) * 16 + v + rlo;
        if (OUT_BF16)
          ((bf16_t*)Cout)[(size_t)row * N + col] = (bf16_t)(acc[mi][ni][v]);
        else
          ((float*)Cout)[(size_t)row * N + col] = acc[mi][ni][v] + bv;
      }
    }
  }
}

// ---------------- fused LSTM step (one layer, one timestep) ----------------
// A = concat(A1[64,lda1], A2[64,lda2]) along K (split at ksplit), z = A * wT^T + b.
// Block owns h-cols [blk*32, +32); computes 4 gate tiles (f,i,o,g), cell update.
__global__ __launch_bounds__(256) void k_lstm_step(
    const bf16_t* __restrict__ A1, int lda1,
    const bf16_t* __restrict__ A2, int lda2,
    int ksplit, int kc,
    const bf16_t* __restrict__ wT,     // [4096][kc], N-major (L2-resident)
    const float* __restrict__ bias,    // [4096]
    float* __restrict__ c_state,       // [64][1024] in/out
    bf16_t* __restrict__ h_next,       // [64][1024] bf16 (ping-pong)
    float* __restrict__ hs_out)        // [64][1024] f32 sequence slot
{
  const int tid = threadIdx.x, lane = tid & 31, wave = tid >> 5;
  const int gate = wave >> 1, nsub = wave & 1;      // 8 waves = 4 gates x 2 halves
  const int hcol0 = blockIdx.x * 32;
  const int ncolb = gate * 1024 + hcol0 + nsub * 16;
  const int lr = lane & 15;
  const int kb = (lane < 16) ? 0 : 8;

  __shared__ __align__(16) bf16_t As[2][64 * 32];
  __shared__ float zbuf[4 * 64 * 32];               // 32 KB

  const int sr = tid >> 2;            // staging row 0..63
  const int skk = (tid & 3) * 8;      // staging k offset {0,8,16,24}

  v8f zero = {0.f, 0.f, 0.f, 0.f, 0.f, 0.f, 0.f, 0.f};
  v8f acc[4];
#pragma unroll
  for (int mt = 0; mt < 4; ++mt) acc[mt] = zero;

  // prologue: tile 0 (k-chunks never straddle ksplit: both are multiples of 32)
  {
    int sk = skk;
    const bf16_t* src = (sk < ksplit) ? (A1 + (size_t)sr * lda1 + sk)
                                      : (A2 + (size_t)sr * lda2 + (sk - ksplit));
    async_load_b128(src, &As[0][sr * 32 + skk]);
  }

  const int nk = kc >> 5;
  for (int ki = 0; ki < nk; ++ki) {
    const int cur = ki & 1;
    wait_async0();
    __syncthreads();
    if (ki + 1 < nk) {
      int sk = ((ki + 1) << 5) + skk;
      const bf16_t* src = (sk < ksplit) ? (A1 + (size_t)sr * lda1 + sk)
                                        : (A2 + (size_t)sr * lda2 + (sk - ksplit));
      async_load_b128(src, &As[cur ^ 1][sr * 32 + skk]);
    }

    // B fragment straight from L2-resident transposed weights
    const bf16_t* bp = wT + (size_t)(ncolb + lr) * kc + (ki << 5) + kb;
    v16bf bfr = mk16(*(const v8bf*)bp, *(const v8bf*)(bp + 16));
    // hoist all 4 A-fragments, then issue WMMAs back-to-back
    v16bf afr[4];
#pragma unroll
    for (int mt = 0; mt < 4; ++mt) {
      const bf16_t* ap = &As[cur][(mt * 16 + lr) * 32 + kb];
      afr[mt] = mk16(*(const v8bf*)ap, *(const v8bf*)(ap + 16));
    }
#pragma unroll
    for (int mt = 0; mt < 4; ++mt)
      acc[mt] = __builtin_amdgcn_wmma_f32_16x16x32_bf16(
          false, afr[mt], false, bfr, (short)0, acc[mt], false, false);

    // pin schedule: B global reads, then all LDS reads, then 4 WMMAs
    SCHED_GROUP(0x020, 2);
    SCHED_GROUP(0x100, 8);
    SCHED_GROUP(0x008, 4);
  }

  float bv = bias[ncolb + lr];
  const int rlo = (lane < 16) ? 0 : 8;
  __syncthreads();
#pragma unroll
  for (int mt = 0; mt < 4; ++mt)
#pragma unroll
    for (int v = 0; v < 8; ++v)
      zbuf[gate * 2048 + (mt * 16 + v + rlo) * 32 + nsub * 16 + lr] = acc[mt][v] + bv;
  __syncthreads();

  for (int i = tid; i < 64 * 32; i += 256) {
    int r = i >> 5, cc = i & 31;
    float f  = sigmoidf_(zbuf[0 * 2048 + r * 32 + cc]);
    float ig = sigmoidf_(zbuf[1 * 2048 + r * 32 + cc]);
    float o  = sigmoidf_(zbuf[2 * 2048 + r * 32 + cc]);
    float g  = tanhf(zbuf[3 * 2048 + r * 32 + cc]);
    int hc = hcol0 + cc;
    float co = c_state[(size_t)r * 1024 + hc];
    float cn = f * co + ig * g;
    float hn = o * tanhf(cn);
    c_state[(size_t)r * 1024 + hc] = cn;
    h_next[(size_t)r * 1024 + hc] = (bf16_t)hn;
    hs_out[(size_t)r * 1024 + hc] = hn;
  }
}

// ---------------- fused output gating: sigmoid(full@gw) blend -> bf16 ----------------
__global__ __launch_bounds__(256) void k_gate(const float* __restrict__ h0s,
                                              const float* __restrict__ h1s,
                                              const float* __restrict__ gw,  // (2048,2)
                                              bf16_t* __restrict__ gated) {
  int row = blockIdx.x, tid = threadIdx.x;
  const float* h0r = h0s + (size_t)row * 1024;
  const float* h1r = h1s + (size_t)row * 1024;
  float p0 = 0.f, p1 = 0.f;
  for (int k = tid; k < 1024; k += 256) {
    float a = h0r[k]; p0 += a * gw[2 * k];              p1 += a * gw[2 * k + 1];
    float b = h1r[k]; p0 += b * gw[2 * (1024 + k)];     p1 += b * gw[2 * (1024 + k) + 1];
  }
  __shared__ float red0[256], red1[256];
  red0[tid] = p0; red1[tid] = p1;
  __syncthreads();
  for (int s = 128; s > 0; s >>= 1) {
    if (tid < s) { red0[tid] += red0[tid + s]; red1[tid] += red1[tid + s]; }
    __syncthreads();
  }
  float g0 = sigmoidf_(red0[0]), g1 = sigmoidf_(red1[0]);
  for (int k = tid; k < 1024; k += 256)
    gated[(size_t)row * 1024 + k] = (bf16_t)(g0 * h0r[k] + g1 * h1r[k]);
}

// ---------------- launcher ----------------
extern "C" void kernel_launch(void* const* d_in, const int* in_sizes, int n_in,
                              void* d_out, int out_size, void* d_ws, size_t ws_size,
                              hipStream_t stream) {
  (void)in_sizes; (void)n_in; (void)out_size; (void)ws_size;
  const float* inputs = (const float*)d_in[0];
  const float* embm   = (const float*)d_in[1];
  const float* w0     = (const float*)d_in[2];
  const float* b0     = (const float*)d_in[3];
  const float* w1     = (const float*)d_in[4];
  const float* b1     = (const float*)d_in[5];
  const float* gw     = (const float*)d_in[6];
  const float* outw   = (const float*)d_in[7];
  const float* outb   = (const float*)d_in[8];
  float* logits = (float*)d_out;

  constexpr int Tt = 64, Bb = 64, V = 32000, E = 512, NH = 1024, G4 = 4096;
  constexpr int TB = Tt * Bb;  // 4096

  char* p = (char*)d_ws;
  auto alloc = [&](size_t bytes) -> void* {
    void* r = (void*)p;
    p += (bytes + 255) & ~(size_t)255;
    return r;
  };
  bf16_t* inputs_bf = (bf16_t*)alloc((size_t)TB * V * 2);       // 262 MB
  bf16_t* embT      = (bf16_t*)alloc((size_t)E * V * 2);        // 33 MB  [E][V]
  bf16_t* w0T       = (bf16_t*)alloc((size_t)G4 * (E + NH) * 2);// 13 MB  [4096][1536]
  bf16_t* w1T       = (bf16_t*)alloc((size_t)G4 * (2 * NH) * 2);// 17 MB  [4096][2048]
  bf16_t* owT       = (bf16_t*)alloc((size_t)V * NH * 2);       // 66 MB  [V][1024]
  bf16_t* embA      = (bf16_t*)alloc((size_t)TB * E * 2);       // 4 MB   [TB][E]
  float*  h0s       = (float*)alloc((size_t)TB * NH * 4);       // 17 MB
  float*  h1s       = (float*)alloc((size_t)TB * NH * 4);       // 17 MB
  bf16_t* gated     = (bf16_t*)alloc((size_t)TB * NH * 2);      // 8 MB
  bf16_t* h0buf     = (bf16_t*)alloc((size_t)2 * Bb * NH * 2);  // ping-pong
  bf16_t* h1buf     = (bf16_t*)alloc((size_t)2 * Bb * NH * 2);
  float*  c0        = (float*)alloc((size_t)Bb * NH * 4);
  float*  c1        = (float*)alloc((size_t)Bb * NH * 4);

  hipMemsetAsync(h0buf, 0, (size_t)2 * Bb * NH * 2, stream);
  hipMemsetAsync(h1buf, 0, (size_t)2 * Bb * NH * 2, stream);
  hipMemsetAsync(c0, 0, (size_t)Bb * NH * 4, stream);
  hipMemsetAsync(c1, 0, (size_t)Bb * NH * 4, stream);

  // weight/input precision prep (once per launch, deterministic)
  k_cvt<<<4096, 256, 0, stream>>>(inputs, inputs_bf, (long long)TB * V);
  dim3 tb(32, 8);
  k_transpose_cvt<<<dim3(E / 32, V / 32), tb, 0, stream>>>(embm, embT, V, E);
  k_transpose_cvt<<<dim3(G4 / 32, (E + NH) / 32), tb, 0, stream>>>(w0, w0T, E + NH, G4);
  k_transpose_cvt<<<dim3(G4 / 32, (2 * NH) / 32), tb, 0, stream>>>(w1, w1T, 2 * NH, G4);
  k_transpose_cvt<<<dim3(V / 32, NH / 32), tb, 0, stream>>>(outw, owT, NH, V);

  // embedding GEMM: embA(TB,E) = inputs_bf(TB,V) x embT(E,V)^T   (memory-bound)
  k_gemm<true, false><<<dim3(E / 128, TB / 128), 256, 0, stream>>>(
      inputs_bf, embT, embA, nullptr, TB, E, V);

  // recurrence: weights stay L2-resident across all 64 steps
  for (int t = 0; t < Tt; ++t) {
    int cur = t & 1;
    k_lstm_step<<<32, 256, 0, stream>>>(
        embA + (size_t)t * Bb * E, E,
        h0buf + (size_t)cur * Bb * NH, NH,
        E, E + NH, w0T, b0, c0,
        h0buf + (size_t)(cur ^ 1) * Bb * NH,
        h0s + (size_t)t * Bb * NH);
    k_lstm_step<<<32, 256, 0, stream>>>(
        h0buf + (size_t)(cur ^ 1) * Bb * NH, NH,
        h1buf + (size_t)cur * Bb * NH, NH,
        NH, 2 * NH, w1T, b1, c1,
        h1buf + (size_t)(cur ^ 1) * Bb * NH,
        h1s + (size_t)t * Bb * NH);
  }

  k_gate<<<TB, 256, 0, stream>>>(h0s, h1s, gw, gated);

  // output projection: logits(TB,V) = gated(TB,NH) x owT(V,NH)^T + out_b
  k_gemm<false, true><<<dim3(V / 128, TB / 128), 256, 0, stream>>>(
      gated, owT, logits, outb, TB, V, NH);
}
